// UpBlock_3006477107937
// MI455X (gfx1250) — matrix operand order repaired and checked by reading, not compiled, not run on hardware
//
#include <hip/hip_runtime.h>

// ---------------------------------------------------------------------------
// Types
// ---------------------------------------------------------------------------
typedef __attribute__((ext_vector_type(16))) __bf16 v16bf;
typedef __attribute__((ext_vector_type(8)))  __bf16 v8bf;
typedef __attribute__((ext_vector_type(4)))  __bf16 v4bf;
typedef __attribute__((ext_vector_type(8)))  float  v8f;
typedef __attribute__((ext_vector_type(4)))  float  v4f;

#define BN_EPS 1e-5f
#define SLOPE  0.01f
#define COUT   64

static __device__ __host__ __forceinline__ __bf16 f2bf(float f) {
    unsigned u = __builtin_bit_cast(unsigned, f);
    u += 0x7FFFu + ((u >> 16) & 1u);            // round-to-nearest-even
    unsigned short s = (unsigned short)(u >> 16);
    return __builtin_bit_cast(__bf16, s);
}

// ---------------------------------------------------------------------------
// Prep kernels: f32 -> bf16 features, and pack weights into WMMA B-fragment
// order.  Packed element index e = ((k*NKC + kc)*4 + ct)*512 + lane*16 + j,
// mapping to W[k][ kc*32 + (lane>=16 ? 8:0) + j + (j&8) ][ ct*16 + lane%16 ].
// (16-bit B 32x16: lanes 0-15 hold K {0..7,16..23}, lanes 16-31 hold K {8..15,24..31}.)
// ---------------------------------------------------------------------------
__global__ void f32_to_bf16_kernel(const float* __restrict__ in,
                                   __bf16* __restrict__ out, long long n) {
    long long i = (long long)blockIdx.x * blockDim.x + threadIdx.x;
    long long stride = (long long)gridDim.x * blockDim.x;
    for (; i < n; i += stride) out[i] = f2bf(in[i]);
}

__global__ void pack_weights_kernel(const float* __restrict__ W,
                                    __bf16* __restrict__ out,
                                    int taps, int cin) {
    const int nkc = cin >> 5;
    const long long total = (long long)taps * nkc * 4 * 512;
    const long long stride = (long long)gridDim.x * blockDim.x;
    for (long long e = (long long)blockIdx.x * blockDim.x + threadIdx.x;
         e < total; e += stride) {
        int j    = (int)(e & 15);
        int lane = (int)((e >> 4) & 31);
        int ct   = (int)((e >> 9) & 3);
        long long t = e >> 11;
        int kc = (int)(t % nkc);
        int k  = (int)(t / nkc);
        int kk = kc * 32 + ((lane & 16) ? 8 : 0) + j + (j & 8);
        int n  = ct * 16 + (lane & 15);
        out[e] = f2bf(W[((long long)k * cin + kk) * COUT + n]);
    }
}

// ---------------------------------------------------------------------------
// Sparse-conv kernel: ONE WAVE computes a 32-point x 64-cout tile
// (two 16x64 WMMA row-tiles -> 8 v8f accumulators). The 4 weight B-fragments
// per K-chunk are amortized over both row-tiles, halving weight-load traffic
// per WMMA vs a 16-point tile.
//   MODE 0: BN + LeakyReLU  -> bf16 out
//   MODE 1: + skip          -> bf16 out
//   MODE 2: BN + LeakyReLU  -> f32  out (final layer)
// Requires npts % 32 == 0 (N = 100000 -> grid 3125, block 32).
// ---------------------------------------------------------------------------
template <int CIN, int TAPS, int MODE>
__global__ __launch_bounds__(32) void sconv_kernel(
    const __bf16* __restrict__ feats,   // [npts, CIN] bf16
    const __bf16* __restrict__ wpk,     // packed weights
    const int*    __restrict__ nbr,     // [TAPS, npts]
    const float*  __restrict__ bnp,     // this layer's [4,64] (g,b,mu,var)
    const float*  __restrict__ skip,    // [npts,64] f32 (MODE 1)
    void*         __restrict__ outv,
    int npts)
{
    constexpr int NKC = CIN / 32;
    __shared__ float smem[32 * 68];

    const int lane  = threadIdx.x & 31;
    const int m16   = lane & 15;
    const int hi    = (lane & 16) ? 1 : 0;
    const int pbase = blockIdx.x * 32;

    v8f c[2][4] = {};

    for (int k = 0; k < TAPS; ++k) {
        const long long kb = (long long)k * npts + pbase;
        const int idx0 = nbr[kb + m16];
        const int idx1 = nbr[kb + 16 + m16];
        // A-fragment rows: lanes<16 hold K {0..7,16..23}, lanes>=16 {8..15,24..31}
        const __bf16* ar0 = feats + (long long)idx0 * CIN + hi * 8;
        const __bf16* ar1 = feats + (long long)idx1 * CIN + hi * 8;
#pragma unroll
        for (int kc = 0; kc < NKC; ++kc) {
            const v8bf* ap0 = (const v8bf*)(ar0 + kc * 32);
            const v8bf* ap1 = (const v8bf*)(ar1 + kc * 32);
            v8bf lo0 = ap0[0], hh0 = ap0[2];
            v8bf lo1 = ap1[0], hh1 = ap1[2];
            v16bf a0, a1;
#pragma unroll
            for (int j = 0; j < 8; ++j) {
                a0[j] = lo0[j]; a0[8 + j] = hh0[j];
                a1[j] = lo1[j]; a1[8 + j] = hh1[j];
            }

            const __bf16* wp = wpk + ((long long)(k * NKC + kc) * 4) * 512 + lane * 16;
#pragma unroll
            for (int ct = 0; ct < 4; ++ct) {
                v16bf b = *(const v16bf*)(wp + ct * 512);
                c[0][ct] = __builtin_amdgcn_wmma_f32_16x16x32_bf16(
                    false, a0, false, b, (short)0, c[0][ct], false, false);
                c[1][ct] = __builtin_amdgcn_wmma_f32_16x16x32_bf16(
                    false, a1, false, b, (short)0, c[1][ct], false, false);
            }
        }
    }

    // Stage f32 tile into LDS: element (m,n), row stride 68 (half-wave writes
    // land 32 banks apart -> conflict-free).
#pragma unroll
    for (int t = 0; t < 2; ++t) {
#pragma unroll
        for (int r = 0; r < 8; ++r) {
            const int m = t * 16 + r + hi * 8;
            smem[m * 68 +  0 + m16] = c[t][0][r];
            smem[m * 68 + 16 + m16] = c[t][1][r];
            smem[m * 68 + 32 + m16] = c[t][2][r];
            smem[m * 68 + 48 + m16] = c[t][3][r];
        }
    }
    __syncthreads();

    // Coalesced readout: lane == point row; each lane walks all 64 couts.
    const int point = pbase + lane;
    const float* lr = &smem[lane * 68];

    if (MODE == 0 || MODE == 2) {
        const float* g  = bnp;
        const float* be = bnp + 64;
        const float* mu = bnp + 128;
        const float* va = bnp + 192;
#pragma unroll
        for (int i = 0; i < 64; i += 4) {
            v4f v = *(const v4f*)(lr + i);
            float y[4];
#pragma unroll
            for (int e = 0; e < 4; ++e) {
                const int cch = i + e;
                float t = (v[e] - mu[cch]) * rsqrtf(va[cch] + BN_EPS) * g[cch] + be[cch];
                y[e] = (t >= 0.f) ? t : SLOPE * t;
            }
            if (MODE == 2) {
                v4f o = { y[0], y[1], y[2], y[3] };
                *(v4f*)((float*)outv + (long long)point * 64 + i) = o;
            } else {
                v4bf o = { f2bf(y[0]), f2bf(y[1]), f2bf(y[2]), f2bf(y[3]) };
                *(v4bf*)((__bf16*)outv + (long long)point * 64 + i) = o;
            }
        }
    } else {  // MODE 1: skip add -> bf16
#pragma unroll
        for (int i = 0; i < 64; i += 4) {
            v4f v = *(const v4f*)(lr + i);
            v4f s = *(const v4f*)(skip + (long long)point * 64 + i);
            v4bf o = { f2bf(v[0] + s[0]), f2bf(v[1] + s[1]),
                       f2bf(v[2] + s[2]), f2bf(v[3] + s[3]) };
            *(v4bf*)((__bf16*)outv + (long long)point * 64 + i) = o;
        }
    }
}

// ---------------------------------------------------------------------------
// Launcher
// ---------------------------------------------------------------------------
extern "C" void kernel_launch(void* const* d_in, const int* in_sizes, int n_in,
                              void* d_out, int out_size, void* d_ws, size_t ws_size,
                              hipStream_t stream) {
    const int N = in_sizes[0] / 128;

    const float* x_feats = (const float*)d_in[0];
    const float* skip    = (const float*)d_in[1];
    const float* W1 = (const float*)d_in[2];
    const float* Wt = (const float*)d_in[3];
    const float* W2 = (const float*)d_in[4];
    const float* W3 = (const float*)d_in[5];
    const float* W4 = (const float*)d_in[6];
    const float* bn = (const float*)d_in[7];
    const int* nbr1 = (const int*)d_in[8];
    const int* nbrt = (const int*)d_in[9];
    const int* nbr2 = (const int*)d_in[10];
    const int* nbr3 = (const int*)d_in[11];
    const int* nbr4 = (const int*)d_in[12];

    char* ws = (char*)d_ws;
    size_t off = 0;
    auto carve = [&](size_t bytes) {
        void* p = ws + off;
        off = (off + bytes + 255) & ~(size_t)255;
        return p;
    };
    __bf16* xb  = (__bf16*)carve((size_t)N * 128 * 2);
    __bf16* hA  = (__bf16*)carve((size_t)N * 64 * 2);
    __bf16* hB  = (__bf16*)carve((size_t)N * 64 * 2);
    __bf16* pW1 = (__bf16*)carve((size_t)27 * 128 * 64 * 2);
    __bf16* pWt = (__bf16*)carve((size_t)27 * 64 * 64 * 2);
    __bf16* pW2 = (__bf16*)carve((size_t)9  * 64 * 64 * 2);
    __bf16* pW3 = (__bf16*)carve((size_t)9  * 64 * 64 * 2);
    __bf16* pW4 = (__bf16*)carve((size_t)27 * 64 * 64 * 2);

    f32_to_bf16_kernel<<<2048, 256, 0, stream>>>(x_feats, xb, (long long)N * 128);
    pack_weights_kernel<<<216, 256, 0, stream>>>(W1, pW1, 27, 128);
    pack_weights_kernel<<<108, 256, 0, stream>>>(Wt, pWt, 27, 64);
    pack_weights_kernel<<< 36, 256, 0, stream>>>(W2, pW2,  9, 64);
    pack_weights_kernel<<< 36, 256, 0, stream>>>(W3, pW3,  9, 64);
    pack_weights_kernel<<<108, 256, 0, stream>>>(W4, pW4, 27, 64);

    const int nblk = N / 32;  // one wave per block, 32-point tile per wave

    // up_block: conv3 (128->64) + BN + LReLU
    sconv_kernel<128, 27, 0><<<nblk, 32, 0, stream>>>(xb, pW1, nbr1, bn + 0 * 256, nullptr, hA, N);
    // transposed conv (64->64), then += skip
    sconv_kernel< 64, 27, 1><<<nblk, 32, 0, stream>>>(hA, pWt, nbrt, nullptr, skip, hB, N);
    // conv_x3
    sconv_kernel< 64,  9, 0><<<nblk, 32, 0, stream>>>(hB, pW2, nbr2, bn + 1 * 256, nullptr, hA, N);
    sconv_kernel< 64,  9, 0><<<nblk, 32, 0, stream>>>(hA, pW3, nbr3, bn + 2 * 256, nullptr, hB, N);
    sconv_kernel< 64, 27, 2><<<nblk, 32, 0, stream>>>(hB, pW4, nbr4, bn + 3 * 256, nullptr, d_out, N);
}
